// LossFunction_85435489452078
// MI455X (gfx1250) — matrix operand order repaired and checked by reading, not compiled, not run on hardware
//
#include <hip/hip_runtime.h>

typedef _Float16 h8   __attribute__((ext_vector_type(8)));
typedef _Float16 v16h __attribute__((ext_vector_type(16)));
typedef float    v8f  __attribute__((ext_vector_type(8)));

#define N_ROWS 8192
#define DIM    192
#define EPSN   1e-8f

#if defined(__has_builtin) && __has_builtin(__builtin_amdgcn_sched_barrier)
#define SCHED_FENCE() __builtin_amdgcn_sched_barrier(0)
#else
#define SCHED_FENCE() do {} while (0)
#endif

// Load two contiguous 8-half (16B) chunks and concatenate into a 16-half WMMA operand.
__device__ __forceinline__ v16h ld_concat(const _Float16* __restrict__ p, int off0, int off1) {
    h8 lo = *(const h8*)(p + off0);
    h8 hi = *(const h8*)(p + off1);
    return __builtin_shufflevector(lo, hi, 0,1,2,3,4,5,6,7,8,9,10,11,12,13,14,15);
}

// ---------------------------------------------------------------------------
// Kernel 1: per-row L2 normalization (fp32), exact fp32 diagonal cosine,
// f16 panels for the WMMA GEMM. One wave32 per row n.
// ---------------------------------------------------------------------------
__global__ void ap_normalize(const float* __restrict__ x,
                             _Float16* __restrict__ pos16,
                             _Float16* __restrict__ anc16,
                             float* __restrict__ diag) {
    const int wid  = threadIdx.x >> 5;
    const int lane = threadIdx.x & 31;
    const int n    = blockIdx.x * 8 + wid;

    const float* __restrict__ xp = x + (size_t)(2 * n) * DIM;
    const float* __restrict__ xa = xp + DIM;

    float p[6], a[6];
    float ssp = 0.f, ssa = 0.f, dp = 0.f;
#pragma unroll
    for (int j = 0; j < 6; ++j) {
        const int idx = lane + 32 * j;        // 6*32 = 192 = DIM
        p[j] = xp[idx];
        a[j] = xa[idx];
        ssp += p[j] * p[j];
        ssa += a[j] * a[j];
        dp  += p[j] * a[j];
    }
#pragma unroll
    for (int m = 16; m >= 1; m >>= 1) {       // full wave32 xor-reduction
        ssp += __shfl_xor(ssp, m);
        ssa += __shfl_xor(ssa, m);
        dp  += __shfl_xor(dp,  m);
    }
    const float sp = 1.f / fmaxf(sqrtf(ssp), EPSN);
    const float sa = 1.f / fmaxf(sqrtf(ssa), EPSN);
    if (lane == 0) diag[n] = dp * sp * sa;    // exact fp32 cos(pos_n, anc_n)

#pragma unroll
    for (int j = 0; j < 6; ++j) {
        const int idx = lane + 32 * j;
        pos16[(size_t)n * DIM + idx] = (_Float16)(p[j] * sp);
        anc16[(size_t)n * DIM + idx] = (_Float16)(a[j] * sa);
    }
}

// ---------------------------------------------------------------------------
// Kernel 2: WMMA GEMM + streaming logsumexp.
// Block = 256 threads = 8 waves; block owns rows [16b, 16b+16).
// Wave w handles column tiles m0 = (w + 8t)*16, t = 0..63.
// Per tile: all 12 global_load_b128 issued as one clause (sched_barrier keeps
// the scheduler from sinking them onto their uses), then 6 back-to-back
// v_wmma_f32_16x16x32_f16 (K = 192) with partial loadcnt waits, then the
// exp epilogue (VALU co-executes with the WMMA pipe across waves).
// rowloss[n] = w*diag[n] - log( sum_m exp(w*c_nm) )   (bias b cancels)
// ---------------------------------------------------------------------------
__global__ void ap_gemm_lse(const _Float16* __restrict__ pos16,
                            const _Float16* __restrict__ anc16,
                            const float* __restrict__ diag,
                            const float* __restrict__ wscale,
                            float* __restrict__ rowloss) {
    __shared__ float lds_sum[16];

    const int tid  = threadIdx.x;
    const int wid  = tid >> 5;
    const int lane = tid & 31;
    const int h    = lane >> 4;     // half-wave select (ISA layout tables)
    const int l15  = lane & 15;

    if (tid < 16) lds_sum[tid] = 0.f;
    __syncthreads();

    const float wv      = wscale[0];
    const int   rowBase = blockIdx.x * 16;

    // A tile: 16 rows x 192 K, register-resident for the whole kernel.
    // A layout (16-bit A 16x32): lane (M=l15, h): halves K = 32k + 16h + e (contiguous 16)
    const _Float16* __restrict__ aptr = pos16 + (size_t)(rowBase + l15) * DIM;
    v16h A[6];
#pragma unroll
    for (int k = 0; k < 6; ++k)
        A[k] = ld_concat(aptr, 32 * k + 16 * h, 32 * k + 16 * h + 8);

    float se[8] = {0.f, 0.f, 0.f, 0.f, 0.f, 0.f, 0.f, 0.f};

    for (int t = 0; t < 64; ++t) {
        const int m0 = (wid + 8 * t) * 16;
        const _Float16* __restrict__ bptr = anc16 + (size_t)(m0 + l15) * DIM;
        if (t + 2 < 64)                       // warm the near caches two tiles ahead
            __builtin_prefetch(anc16 + (size_t)((wid + 8 * (t + 2)) * 16 + l15) * DIM, 0, 3);

        // B layout (16-bit B 32x16): lane (N=l15, h): halves K = 16h + e (contiguous 16)
        v16h B[6];
#pragma unroll
        for (int k = 0; k < 6; ++k)
            B[k] = ld_concat(bptr, 32 * k + 16 * h, 32 * k + 16 * h + 8);

        // Keep all 12 loads issued before the first WMMA: one clause, partial waits.
        SCHED_FENCE();

        v8f c = {};
#pragma unroll
        for (int k = 0; k < 6; ++k)
            c = __builtin_amdgcn_wmma_f32_16x16x32_f16(
                    false, A[k], false, B[k], (short)0, c, false, false);

        // C layout: VGPR r holds row M = r + 8h, column N = m0 + l15
#pragma unroll
        for (int r = 0; r < 8; ++r)
            se[r] += __expf(wv * c[r]);       // v_mul + v_mul + v_exp_f32
    }

    // Reduce the 16 lanes that share the same h (same 8 rows: M = r + 8h).
#pragma unroll
    for (int m = 8; m >= 1; m >>= 1) {
#pragma unroll
        for (int r = 0; r < 8; ++r)
            se[r] += __shfl_xor(se[r], m);
    }
    if (l15 == 0) {
#pragma unroll
        for (int r = 0; r < 8; ++r)
            atomicAdd(&lds_sum[r + 8 * h], se[r]);   // ds_add_f32 across 8 waves
    }
    __syncthreads();

    if (tid < 16) {
        const int n = rowBase + tid;
        rowloss[n] = wv * diag[n] - logf(lds_sum[tid]);
    }
}

// ---------------------------------------------------------------------------
// Kernel 3: deterministic tree reduction of row losses -> scalar loss.
// ---------------------------------------------------------------------------
__global__ void ap_finalize(const float* __restrict__ rowloss,
                            float* __restrict__ out) {
    __shared__ float red[256];
    float s = 0.f;
    for (int i = threadIdx.x; i < N_ROWS; i += 256) s += rowloss[i];
    red[threadIdx.x] = s;
    __syncthreads();
    for (int m = 128; m > 0; m >>= 1) {
        if (threadIdx.x < m) red[threadIdx.x] += red[threadIdx.x + m];
        __syncthreads();
    }
    if (threadIdx.x == 0) out[0] = -red[0] / (float)N_ROWS;
}

// ---------------------------------------------------------------------------
extern "C" void kernel_launch(void* const* d_in, const int* in_sizes, int n_in,
                              void* d_out, int out_size, void* d_ws, size_t ws_size,
                              hipStream_t stream) {
    const float* x = (const float*)d_in[0];
    const float* w = (const float*)d_in[1];
    // d_in[2] is b; it cancels exactly in diag(log_softmax) so it is unused.
    (void)in_sizes; (void)n_in; (void)out_size; (void)ws_size;

    char* ws = (char*)d_ws;
    float*    diag    = (float*)(ws);                                        //  32 KB
    float*    rowloss = (float*)(ws + 32768);                                //  32 KB
    _Float16* pos16   = (_Float16*)(ws + 65536);                             //   3 MB
    _Float16* anc16   = (_Float16*)(ws + 65536 + (size_t)N_ROWS * DIM * 2);  //   3 MB

    ap_normalize<<<N_ROWS / 8,  256, 0, stream>>>(x, pos16, anc16, diag);
    ap_gemm_lse <<<N_ROWS / 16, 256, 0, stream>>>(pos16, anc16, diag, w, rowloss);
    ap_finalize <<<1,           256, 0, stream>>>(rowloss, (float*)d_out);
}
